// MagicGraphAttention_32633161515918
// MI455X (gfx1250) — matrix operand order repaired and checked by reading, not compiled, not run on hardware
//
#include <hip/hip_runtime.h>
#include <hip/hip_bf16.h>

// Problem dims (from reference)
#define BS     32
#define NN     512
#define IN_F   256
#define OUT_F  128
#define HEADS  4
#define HD     (HEADS * OUT_F)   // 512
#define NEG_SLOPE 0.2f

typedef __attribute__((ext_vector_type(16))) __bf16 v16bf;
typedef __attribute__((ext_vector_type(8)))  __bf16 v8bf;
typedef __attribute__((ext_vector_type(8)))  float  v8f;
typedef __attribute__((ext_vector_type(4)))  unsigned int uint32x4;
typedef __attribute__((ext_vector_type(4)))  int  int32x4;
typedef __attribute__((ext_vector_type(8)))  int  int32x8;

// f32 -> bf16 bits, round-to-nearest-even (ushort storage: no host __bf16 deps)
__device__ __forceinline__ unsigned short f2bfu(float f) {
    unsigned u = __builtin_bit_cast(unsigned, f);
    u += 0x7FFFu + ((u >> 16) & 1u);
    return (unsigned short)(u >> 16);
}

// ---------------------------------------------------------------------------
// TDM: async 2D bf16 tile load (tile1 rows x tile0 cols, row stride stride0
// elements) from global into LDS, per CDNA5 D# layout (ISA 08_async_tensor §8).
// Issued once per wave; tracked with TENSORcnt.
// ---------------------------------------------------------------------------
__device__ __forceinline__ void tdm_load_tile_bf16(
    unsigned short* lds_dst, const unsigned short* gsrc,
    unsigned dim0, unsigned dim1, unsigned stride0,
    unsigned tile0, unsigned tile1)
{
    const unsigned lds_addr = (unsigned)(unsigned long long)(void*)lds_dst;
    const unsigned long long ga = (unsigned long long)gsrc;

    uint32x4 g0 = {};
    g0[0] = 1u;                                            // count=1 (valid), user D#
    g0[1] = lds_addr;                                      // lds_addr[31:0]
    g0[2] = (unsigned)(ga & 0xFFFFFFFFull);                // global_addr[31:0]
    g0[3] = (unsigned)((ga >> 32) & 0x1FFFFFFull)          // global_addr[56:32]
          | (2u << 30);                                    // type = 2 ("image")

    int32x8 g1 = {};
    g1[0] = (int)(1u << 16);                               // data_size = 2 bytes
    g1[1] = (int)((dim0 & 0xFFFFu) << 16);                 // tensor_dim0[15:0] @bit48
    g1[2] = (int)(((dim0 >> 16) & 0xFFFFu) | ((dim1 & 0xFFFFu) << 16));
    g1[3] = (int)(((dim1 >> 16) & 0xFFFFu) | ((tile0 & 0xFFFFu) << 16)); // tile_dim0
    g1[4] = (int)(tile1 & 0xFFFFu);                        // tile_dim1 (tile_dim2=0)
    g1[5] = (int)stride0;                                  // tensor_dim0_stride[31:0]
    g1[6] = 0;                                             // stride0 hi / stride1 lo
    g1[7] = 0;

    int32x4 g2 = {};
    int32x4 g3 = {};
#if __clang_major__ >= 23
    int32x8 g4 = {};
    __builtin_amdgcn_tensor_load_to_lds(g0, g1, g2, g3, g4, 0);
#else
    __builtin_amdgcn_tensor_load_to_lds(g0, g1, g2, g3, 0);
#endif
}

// ---------------------------------------------------------------------------
// Kernel 0 (prep): X -> bf16; W -> bf16 in WMMA B-fragment-linear order:
//   Wf[(((ntile*8 + ktile)*32 + lane)*16) + i]  where for W element (k, c):
//   ktile=k>>5, lh=(k&31)>>4, i=k&15, ntile=c>>4, ln=c&15, lane=lh*16+ln.
// One pass; GEMM 1 then needs only contiguous 32B loads per lane.
// ---------------------------------------------------------------------------
__global__ __launch_bounds__(256) void gat_prep(
    const float* __restrict__ X, const float* __restrict__ Wm,
    unsigned short* __restrict__ Xb, unsigned short* __restrict__ Wf)
{
    const int t      = blockIdx.x * 256 + threadIdx.x;
    const int stride = gridDim.x * 256;
    for (int i = t; i < BS * NN * IN_F; i += stride) Xb[i] = f2bfu(X[i]);
    for (int i = t; i < IN_F * HD; i += stride) {
        const int k = i >> 9, c = i & 511;
        const int ktile = k >> 5, kl = k & 31, lh = kl >> 4, ii = kl & 15;
        const int ntile = c >> 4, ln = c & 15, lane = lh * 16 + ln;
        Wf[(((size_t)(ntile * 8 + ktile) * 32 + lane) * 16) + ii] = f2bfu(Wm[i]);
    }
}

// ---------------------------------------------------------------------------
// Kernel 1: h = X @ W (M=16384,K=256,N=512) bf16 WMMA.
// A tile (16x32) via double-buffered TDM into LDS; B via fragment-linear Wf.
// Writes h f32 [b,n,HD] (for coeff kernel) and h bf16 in fragment-linear
// order for kernel 4's B operand: hbF[(((bh*16+ktile)*8+ntile)*32+lane)*16+i].
// ---------------------------------------------------------------------------
__global__ __launch_bounds__(256) void gat_proj_wmma(
    const unsigned short* __restrict__ Xb, const unsigned short* __restrict__ Wf,
    float* __restrict__ hf, unsigned short* __restrict__ hbF)
{
    __shared__ __align__(16) unsigned short As[2][16 * 32];
    const int tid  = threadIdx.x;
    const int lane = tid & 31;
    const int warp = tid >> 5;
    const int lh   = lane >> 4;
    const int ln   = lane & 15;

    const int mtile = blockIdx.x >> 2;               // 0..1023
    const int ntile = (blockIdx.x & 3) * 8 + warp;   // 0..31
    const int rowb  = mtile * 16;
    const int colb  = ntile * 16;

    if (warp == 0)   // first A tile
        tdm_load_tile_bf16(&As[0][0], Xb + (size_t)rowb * IN_F,
                           IN_F, 16, IN_F, 32, 16);

    v8f c = {};
    int buf = 0;
    for (int k0 = 0; k0 < IN_F; k0 += 32) {
        const bool nxt = (k0 + 32) < IN_F;
        if (warp == 0) {
            if (nxt)
                tdm_load_tile_bf16(&As[buf ^ 1][0],
                                   Xb + (size_t)rowb * IN_F + k0 + 32,
                                   IN_F, 16, IN_F, 32, 16);
            if (nxt) __builtin_amdgcn_s_wait_tensorcnt(1);
            else     __builtin_amdgcn_s_wait_tensorcnt(0);
        }
        __syncthreads();

        // A fragment: 16-bit A 16x32 layout = two contiguous 8-short runs/lane
        const v8bf alo = *(const v8bf*)&As[buf][ln * 32 + 16 * lh];
        const v8bf ahi = *(const v8bf*)&As[buf][ln * 32 + 16 * lh + 8];
        v16bf a;
#pragma unroll
        for (int i = 0; i < 8; ++i) { a[i] = alo[i]; a[i + 8] = ahi[i]; }

        // B fragment: contiguous 32B per lane from fragment-linear Wf
        const int ktile = k0 >> 5;
        const unsigned short* bp =
            Wf + ((size_t)(ntile * 8 + ktile) * 32 + lane) * 16;
        if (nxt) __builtin_prefetch(bp + 512, 0, 3);  // next ktile, same ntile
        const v8bf blo = *(const v8bf*)(bp);
        const v8bf bhi = *(const v8bf*)(bp + 8);
        v16bf b;
#pragma unroll
        for (int i = 0; i < 8; ++i) { b[i] = blo[i]; b[i + 8] = bhi[i]; }

        c = __builtin_amdgcn_wmma_f32_16x16x32_bf16(false, a, false, b,
                                                    (short)0, c, false, false);
        __syncthreads();
        buf ^= 1;
    }

    // C/D layout: VGPR r -> M = r + 8*half, N = lane&15
#pragma unroll
    for (int r = 0; r < 8; ++r) {
        const int row = rowb + r + 8 * lh;           // b*NN + node
        const int col = colb + ln;                   // head*128 + d
        hf[(size_t)row * HD + col] = c[r];
        const int bb = row >> 9, n = row & 511;
        const int hh = col >> 7, d = col & 127;
        // fragment coords for kernel 4 (K dim = node, N dim = d):
        const int kt4 = n >> 5, r5 = n & 31, lh4 = r5 >> 4, i4 = r5 & 15;
        const int nt4 = d >> 4, ln4 = d & 15, lane4 = lh4 * 16 + ln4;
        hbF[(((size_t)((bb * HEADS + hh) * 16 + kt4) * 8 + nt4) * 32 + lane4) * 16 + i4]
            = f2bfu(c[r]);
    }
}

// ---------------------------------------------------------------------------
// Kernel 2: coeff_i/j[b,h,n] = h[b,n,h,:] . a_i/j[h,:]  (wave per (b,n,h))
// ---------------------------------------------------------------------------
__global__ __launch_bounds__(256) void gat_coeff(
    const float* __restrict__ hf, const float* __restrict__ ai,
    const float* __restrict__ aj, float* __restrict__ ci, float* __restrict__ cj)
{
    const int wid  = blockIdx.x * 8 + (threadIdx.x >> 5);  // 0..65535
    const int lane = threadIdx.x & 31;
    const int b    = wid >> 11;
    const int rem  = wid & 2047;
    const int n    = rem >> 2;
    const int h    = rem & 3;

    const float* hrow = hf + ((size_t)(b * NN + n) * HD + h * OUT_F);
    const float* aih  = ai + h * OUT_F;
    const float* ajh  = aj + h * OUT_F;

    float si = 0.f, sj = 0.f;
    for (int d = lane; d < OUT_F; d += 32) {
        const float v = hrow[d];
        si += v * aih[d];
        sj += v * ajh[d];
    }
#pragma unroll
    for (int off = 16; off >= 1; off >>= 1) {
        si += __shfl_xor(si, off, 32);
        sj += __shfl_xor(sj, off, 32);
    }
    if (lane == 0) {
        ci[(size_t)(b * HEADS + h) * NN + n] = si;
        cj[(size_t)(b * HEADS + h) * NN + n] = sj;
    }
}

// ---------------------------------------------------------------------------
// Kernel 3: masked-softmax attention rows -> bf16 att[b,h,i,j]
// One wave per (b,h,i); 16 j-values/lane in registers, wave32 reductions.
// ---------------------------------------------------------------------------
__global__ __launch_bounds__(256) void gat_attn_rows(
    const float* __restrict__ adj, const float* __restrict__ ci,
    const float* __restrict__ cj, unsigned short* __restrict__ attb)
{
    const int wid  = blockIdx.x * 8 + (threadIdx.x >> 5);  // 0..65535
    const int lane = threadIdx.x & 31;
    const int b    = wid >> 11;
    const int rem  = wid & 2047;
    const int h    = rem >> 9;
    const int i    = rem & 511;

    const float  civ  = ci[(size_t)(b * HEADS + h) * NN + i];
    const float* adjr = adj + (size_t)(b * NN + i) * NN;
    const float* cjr  = cj + (size_t)(b * HEADS + h) * NN;

    float vals[16];
    float mx = -3.0e38f;
#pragma unroll
    for (int t = 0; t < 16; ++t) {
        const int j = t * 32 + lane;
        float e = civ + cjr[j];
        e = (e > 0.f) ? e : NEG_SLOPE * e;                 // leaky relu
        const float m = (adjr[j] > 0.f) ? e : -1.0e9f;     // mask
        vals[t] = m;
        mx = fmaxf(mx, m);
    }
#pragma unroll
    for (int off = 16; off >= 1; off >>= 1) mx = fmaxf(mx, __shfl_xor(mx, off, 32));

    float s = 0.f;
#pragma unroll
    for (int t = 0; t < 16; ++t) {
        const float p = __expf(vals[t] - mx);  // masked -> exp(-1e9-mx) == 0
        vals[t] = p;
        s += p;
    }
#pragma unroll
    for (int off = 16; off >= 1; off >>= 1) s += __shfl_xor(s, off, 32);
    const float inv = 1.0f / s;

    unsigned short* orow = attb + ((size_t)(b * HEADS + h) * NN + i) * NN;
#pragma unroll
    for (int t = 0; t < 16; ++t) orow[t * 32 + lane] = f2bfu(vals[t] * inv);
}

// ---------------------------------------------------------------------------
// Kernel 4: out[b,i,h*128+d] = att[b,h](512x512) @ h_bf16[b,h](512x128) + bias
// A tile via double-buffered TDM; B from fragment-linear hbF (32B/lane loads).
// Block per (b,h,mtile); 8 waves cover full D=128.
// ---------------------------------------------------------------------------
__global__ __launch_bounds__(256) void gat_aggr_wmma(
    const unsigned short* __restrict__ attb, const unsigned short* __restrict__ hbF,
    const float* __restrict__ bias, float* __restrict__ out)
{
    __shared__ __align__(16) unsigned short As[2][16 * 32];
    const int tid  = threadIdx.x;
    const int lane = tid & 31;
    const int warp = tid >> 5;
    const int lh   = lane >> 4;
    const int ln   = lane & 15;

    const int b     = blockIdx.x >> 7;
    const int rem   = blockIdx.x & 127;
    const int h     = rem >> 5;
    const int mtile = rem & 31;
    const int bh    = b * HEADS + h;

    const unsigned short* Abase = attb + ((size_t)bh * NN + mtile * 16) * NN;
    const int colb = warp * 16;                          // d within head

    if (warp == 0)
        tdm_load_tile_bf16(&As[0][0], Abase, NN, 16, NN, 32, 16);

    v8f c = {};
    int buf = 0;
    for (int k0 = 0; k0 < NN; k0 += 32) {
        const bool nxt = (k0 + 32) < NN;
        if (warp == 0) {
            if (nxt)
                tdm_load_tile_bf16(&As[buf ^ 1][0], Abase + k0 + 32,
                                   NN, 16, NN, 32, 16);
            if (nxt) __builtin_amdgcn_s_wait_tensorcnt(1);
            else     __builtin_amdgcn_s_wait_tensorcnt(0);
        }
        __syncthreads();

        const v8bf alo = *(const v8bf*)&As[buf][ln * 32 + 16 * lh];
        const v8bf ahi = *(const v8bf*)&As[buf][ln * 32 + 16 * lh + 8];
        v16bf a;
#pragma unroll
        for (int i = 0; i < 8; ++i) { a[i] = alo[i]; a[i + 8] = ahi[i]; }

        const int ktile = k0 >> 5;
        const unsigned short* bp =
            hbF + (((size_t)(bh * 16 + ktile) * 8 + warp) * 32 + lane) * 16;
        if (nxt) __builtin_prefetch(bp + 8 * 32 * 16, 0, 3);  // next ktile
        const v8bf blo = *(const v8bf*)(bp);
        const v8bf bhi = *(const v8bf*)(bp + 8);
        v16bf bm;
#pragma unroll
        for (int i = 0; i < 8; ++i) { bm[i] = blo[i]; bm[i + 8] = bhi[i]; }

        c = __builtin_amdgcn_wmma_f32_16x16x32_bf16(false, a, false, bm,
                                                    (short)0, c, false, false);
        __syncthreads();
        buf ^= 1;
    }

#pragma unroll
    for (int r = 0; r < 8; ++r) {
        const int row = mtile * 16 + r + 8 * lh;  // node i
        const int col = colb + ln;                // d within head
        out[(size_t)(b * NN + row) * HD + h * OUT_F + col] = c[r] + bias[h * OUT_F + col];
    }
}

// ---------------------------------------------------------------------------
// Host-side launcher
// ---------------------------------------------------------------------------
extern "C" void kernel_launch(void* const* d_in, const int* in_sizes, int n_in,
                              void* d_out, int out_size, void* d_ws, size_t ws_size,
                              hipStream_t stream) {
    const float* X    = (const float*)d_in[0];   // [32,512,256]
    const float* adj  = (const float*)d_in[1];   // [32,512,512]
    const float* Wm   = (const float*)d_in[2];   // [256,512]
    const float* ai   = (const float*)d_in[3];   // [4,128,1]
    const float* aj   = (const float*)d_in[4];   // [4,128,1]
    const float* bias = (const float*)d_in[5];   // [512]
    float* out = (float*)d_out;                  // [32,512,512]

    char* ws = (char*)d_ws;
    // Workspace layout (bytes):
    float*          hf   = (float*)(ws);                      //  33,554,432 : h f32 [b,n,HD]
    unsigned short* hbF  = (unsigned short*)(ws + 33554432);  //  16,777,216 : h bf16, frag order
    float*          ci   = (float*)(ws + 50331648);           //     262,144 : coeff_i [b,h,n]
    float*          cj   = (float*)(ws + 50593792);           //     262,144 : coeff_j [b,h,n]
    unsigned short* attb = (unsigned short*)(ws + 50855936);  //  67,108,864 : att bf16 [b,h,i,j]
    unsigned short* Xb   = (unsigned short*)(ws + 117964800); //   8,388,608 : X bf16
    unsigned short* Wf   = (unsigned short*)(ws + 126353408); //     262,144 : W bf16, frag order
    // total ~120.7 MB

    gat_prep<<<dim3(2048), dim3(256), 0, stream>>>(X, Wm, Xb, Wf);
    gat_proj_wmma<<<dim3(4096), dim3(256), 0, stream>>>(Xb, Wf, hf, hbF);
    gat_coeff<<<dim3(8192), dim3(256), 0, stream>>>(hf, ai, aj, ci, cj);
    gat_attn_rows<<<dim3(8192), dim3(256), 0, stream>>>(adj, ci, cj, attb);
    gat_aggr_wmma<<<dim3(4096), dim3(256), 0, stream>>>(attb, hbF, bias, out);
}
